// AttentionFreeTransformerEncoder_46110768889965
// MI455X (gfx1250) — compile-verified, matrix operand order checked by reading, not run on hardware
//
#include <hip/hip_runtime.h>
#include <hip/hip_bf16.h>

// ---------------------------------------------------------------------------
// AFT-Full transformer block for MI455X (gfx1250, wave32, WMMA).
// All GEMMs run as bf16 x bf16 -> f32 via V_WMMA_F32_16X16X32_BF16.
// Tile staging uses async global->LDS (ASYNCcnt) with double buffering when
// the toolchain exposes the gfx1250 builtins; sync staging otherwise.
// ---------------------------------------------------------------------------

typedef __attribute__((ext_vector_type(8)))  float  f32x8;
typedef __attribute__((ext_vector_type(8)))  __bf16 bf16x8;
typedef __attribute__((ext_vector_type(16))) __bf16 bf16x16;

#if defined(__HIP_DEVICE_COMPILE__) && defined(__has_builtin)
#if __has_builtin(__builtin_amdgcn_global_load_async_to_lds_b128) && \
    __has_builtin(__builtin_amdgcn_s_wait_asynccnt)
#define AFT_ASYNC 1
#endif
#endif
#ifndef AFT_ASYNC
#define AFT_ASYNC 0
#endif

#if AFT_ASYNC
// Parameter types per hipcc diagnostic: gcc-vector int4 pointers, global (AS1)
// source / LDS (AS3) destination, 4 args total: (gptr, lptr, imm_off, cpol).
typedef int aft_v4i __attribute__((vector_size(16)));
typedef __attribute__((address_space(1))) aft_v4i* aft_gp;
typedef __attribute__((address_space(3))) aft_v4i* aft_lp;
#endif

__device__ __forceinline__ float gelu_exact(float x) {
    return 0.5f * x * (1.0f + erff(x * 0.70710678118654752f));
}

// ---------------------------------------------------------------------------
// Generic batched GEMM:  C[b] = act(A[b] * Bt[b]^T + bias) + res
//   A  : [M,K] bf16 row-major (lda), batch stride sA (elements)
//   Bt : [N,K] bf16 row-major (ldb)  (i.e. B pre-transposed, K-major)
//   C  : optional f32 out  (ldc), Cb : optional bf16 out (ldc)
//   res: optional f32 residual added AFTER activation
// Block: 256 threads = 8 waves. Block tile 128x128, wave tile 64x32, BK=64
// (two WMMA K-steps per staged tile), double-buffered LDS.
// M,N multiples of 128; K multiple of 64 (true for all call sites).
// ---------------------------------------------------------------------------
#define BM 128
#define BN 128
#define BK 64
#define LDS_K (BK + 8)   // pad to break LDS bank conflicts

__global__ __launch_bounds__(256)
void k_gemm_bf16(const __bf16* __restrict__ A,  long long sA, int lda,
                 const __bf16* __restrict__ Bt, long long sB, int ldb,
                 const float*  __restrict__ bias,
                 float*        __restrict__ C,  long long sC,
                 const float*  __restrict__ res, long long sR,
                 __bf16*       __restrict__ Cb, long long sCb,
                 int ldc, int K, int act)
{
    __shared__ __attribute__((aligned(16))) __bf16 As[2][BM * LDS_K];
    __shared__ __attribute__((aligned(16))) __bf16 Bs[2][BN * LDS_K];

    const int bz   = blockIdx.z;
    const long long aB = (long long)bz * sA;
    const long long bB = (long long)bz * sB;
    const int row0 = blockIdx.y * BM;
    const int col0 = blockIdx.x * BN;

    const int tid  = threadIdx.x;
    const int wave = tid >> 5;
    const int lane = tid & 31;
    const int wm   = (wave >> 2) * 64;   // wave M offset within block tile
    const int wn   = (wave & 3)  * 32;   // wave N offset within block tile
    const int lr   = lane & 15;          // row within 16x16 fragment
    const int khi  = lane >> 4;          // K-halfword group select (ISA layout)

    // staging coords: BMxBK bf16 = 16KB per operand; 256 threads x 4 chunks
    const int gc = (tid & 7) * 8;        // bf16 column chunk {0,8,...,56}
    const int gr = tid >> 3;             // row 0..31 (x4 passes)

    auto stage = [&](int buf, int k0) {
#pragma unroll
        for (int p = 0; p < 4; ++p) {
            const int rr = gr + p * 32;
            const __bf16* ag = A  + aB + (long long)(row0 + rr) * lda + k0 + gc;
            const __bf16* bg = Bt + bB + (long long)(col0 + rr) * ldb + k0 + gc;
            __bf16* al = &As[buf][rr * LDS_K + gc];
            __bf16* bl = &Bs[buf][rr * LDS_K + gc];
#if AFT_ASYNC
            __builtin_amdgcn_global_load_async_to_lds_b128(
                (aft_gp)ag, (aft_lp)al, 0, 0);
            __builtin_amdgcn_global_load_async_to_lds_b128(
                (aft_gp)bg, (aft_lp)bl, 0, 0);
#else
            *(bf16x8*)al = *(const bf16x8*)ag;
            *(bf16x8*)bl = *(const bf16x8*)bg;
#endif
        }
    };

    f32x8 acc[4][2];
#pragma unroll
    for (int i = 0; i < 4; ++i)
#pragma unroll
        for (int j = 0; j < 2; ++j)
#pragma unroll
            for (int v = 0; v < 8; ++v) acc[i][j][v] = 0.0f;

    const int nTiles = K / BK;
    stage(0, 0);

    for (int t = 0; t < nTiles; ++t) {
        const int  cur     = t & 1;
        const bool hasNext = (t + 1) < nTiles;
        if (hasNext) stage(cur ^ 1, (t + 1) * BK);   // prefetch next tile
#if AFT_ASYNC
        if (hasNext) __builtin_amdgcn_s_wait_asynccnt(8);  // tile t landed
        else         __builtin_amdgcn_s_wait_asynccnt(0);
#endif
        __syncthreads();

        // two WMMA K-steps over the 64-wide staged tile
#pragma unroll
        for (int ks = 0; ks < 2; ++ks) {
            const int kb = ks * 32;
            // 16-bit 16x32 fragment layout: lanes 0-15 hold K 0..7 & 16..23,
            // lanes 16-31 hold K 8..15 & 24..31 (two ds_load_b128 each).
            bf16x16 af[4], bfv[2];
#pragma unroll
            for (int mt = 0; mt < 4; ++mt) {
                const __bf16* p = &As[cur][(wm + mt * 16 + lr) * LDS_K + kb + khi * 8];
                bf16x8 lo = *(const bf16x8*)p;
                bf16x8 hh = *(const bf16x8*)(p + 16);
#pragma unroll
                for (int e = 0; e < 8; ++e) { af[mt][e] = lo[e]; af[mt][e + 8] = hh[e]; }
            }
#pragma unroll
            for (int nt = 0; nt < 2; ++nt) {
                const __bf16* p = &Bs[cur][(wn + nt * 16 + lr) * LDS_K + kb + khi * 8];
                bf16x8 lo = *(const bf16x8*)p;
                bf16x8 hh = *(const bf16x8*)(p + 16);
#pragma unroll
                for (int e = 0; e < 8; ++e) { bfv[nt][e] = lo[e]; bfv[nt][e + 8] = hh[e]; }
            }
#pragma unroll
            for (int mt = 0; mt < 4; ++mt)
#pragma unroll
                for (int nt = 0; nt < 2; ++nt)
                    acc[mt][nt] = __builtin_amdgcn_wmma_f32_16x16x32_bf16(
                        false, af[mt], false, bfv[nt], (short)0, acc[mt][nt],
                        false, false);
        }
        __syncthreads();
    }

    // Epilogue. C/D layout: VGPR v -> M = v + 8*khi (lanes 0-15 / 16-31), N = lr.
#pragma unroll
    for (int mt = 0; mt < 4; ++mt) {
#pragma unroll
        for (int nt = 0; nt < 2; ++nt) {
#pragma unroll
            for (int v = 0; v < 8; ++v) {
                const int m = row0 + wm + mt * 16 + khi * 8 + v;
                const int n = col0 + wn + nt * 16 + lr;
                float xv = acc[mt][nt][v];
                if (bias) xv += bias[n];
                if (act == 1) xv = gelu_exact(xv);
                const long long idx = (long long)m * ldc + n;
                if (res) xv += res[(long long)bz * sR + idx];
                if (C)  C[(long long)bz * sC + idx]  = xv;
                if (Cb) Cb[(long long)bz * sCb + idx] = (__bf16)xv;
            }
        }
    }
}

// ---------------------------------------------------------------------------
// Weight prep: f32 [K,N] -> bf16 [N,K] (transpose so GEMM B-operand is K-major)
// ---------------------------------------------------------------------------
__global__ __launch_bounds__(256)
void k_wt_bf16(const float* __restrict__ W, __bf16* __restrict__ Wt,
               int Kd, int Nd)
{
    long long i = (long long)blockIdx.x * 256 + threadIdx.x;
    if (i >= (long long)Kd * Nd) return;
    int k = (int)(i / Nd), n = (int)(i % Nd);
    Wt[(long long)n * Kd + k] = (__bf16)W[i];
}

// exp_w = exp(w_pos - rowmax(w_pos)) -> bf16 [T,T]
__global__ __launch_bounds__(256)
void k_expw(const float* __restrict__ W, __bf16* __restrict__ E, int T)
{
    __shared__ float red[256];
    const int t = blockIdx.x;
    const float* row = W + (long long)t * T;
    float mx = -3.4e38f;
    for (int i = threadIdx.x; i < T; i += 256) mx = fmaxf(mx, row[i]);
    red[threadIdx.x] = mx; __syncthreads();
    for (int s = 128; s > 0; s >>= 1) {
        if (threadIdx.x < s) red[threadIdx.x] = fmaxf(red[threadIdx.x], red[threadIdx.x + s]);
        __syncthreads();
    }
    mx = red[0];
    for (int i = threadIdx.x; i < T; i += 256)
        E[(long long)t * T + i] = (__bf16)expf(row[i] - mx);
}

// LayerNorm over F, output bf16 (the only consumer is a bf16 GEMM A-operand)
__global__ __launch_bounds__(256)
void k_ln(const float* __restrict__ X, const float* __restrict__ g,
          const float* __restrict__ b, __bf16* __restrict__ Y, int F)
{
    __shared__ float rs[256], rq[256];
    const long long r = blockIdx.x;
    const float* x = X + r * F;
    float s = 0.f, q = 0.f;
    for (int i = threadIdx.x; i < F; i += 256) { float v = x[i]; s += v; q += v * v; }
    rs[threadIdx.x] = s; rq[threadIdx.x] = q; __syncthreads();
    for (int st = 128; st > 0; st >>= 1) {
        if (threadIdx.x < st) {
            rs[threadIdx.x] += rs[threadIdx.x + st];
            rq[threadIdx.x] += rq[threadIdx.x + st];
        }
        __syncthreads();
    }
    const float mu  = rs[0] / F;
    const float var = rq[0] / F - mu * mu;
    const float inv = rsqrtf(var + 1e-5f);
    for (int i = threadIdx.x; i < F; i += 256)
        Y[r * F + i] = (__bf16)((x[i] - mu) * inv * g[i] + b[i]);
}

// QKV epilogue: rowmax(K) -> exp_K; write [exp_K*V ; exp_K] TRANSPOSED as the
// attention GEMM B-operand attB[b][n][s] (n in [0,2F)); sigmoid(Q) in place.
__global__ __launch_bounds__(256)
void k_qkv_epi(float* __restrict__ Q, const float* __restrict__ K,
               const float* __restrict__ V, __bf16* __restrict__ attB,
               int F, int T)
{
    __shared__ float red[256];
    const int r = blockIdx.x;            // r = b*T + t
    const int b = r / T, t = r % T;
    const float* Kr = K + (long long)r * F;
    float mx = -3.4e38f;
    for (int i = threadIdx.x; i < F; i += 256) mx = fmaxf(mx, Kr[i]);
    red[threadIdx.x] = mx; __syncthreads();
    for (int s = 128; s > 0; s >>= 1) {
        if (threadIdx.x < s) red[threadIdx.x] = fmaxf(red[threadIdx.x], red[threadIdx.x + s]);
        __syncthreads();
    }
    mx = red[0];
    __bf16* base = attB + (long long)b * (2LL * F) * T;
    for (int i = threadIdx.x; i < F; i += 256) {
        float ek = expf(Kr[i] - mx);
        float v  = V[(long long)r * F + i];
        base[(long long)i * T + t]       = (__bf16)(ek * v);   // num columns
        base[(long long)(F + i) * T + t] = (__bf16)ek;         // den columns
        float qv = Q[(long long)r * F + i];
        Q[(long long)r * F + i] = 1.0f / (1.0f + expf(-qv));   // sigQ in place
    }
}

// Yt = sigQ * num/den  (bf16, feeds out-projection GEMM)
__global__ __launch_bounds__(256)
void k_yt(const float* __restrict__ sigQ, const float* __restrict__ attC,
          __bf16* __restrict__ Yt, int F, int T)
{
    const int r = blockIdx.x;
    const int b = r / T, t = r % T;
    const float* row = attC + (long long)b * T * (2LL * F) + (long long)t * (2LL * F);
    for (int i = threadIdx.x; i < F; i += 256) {
        float num = row[i], den = row[F + i];
        Yt[(long long)r * F + i] = (__bf16)(sigQ[(long long)r * F + i] * (num / den));
    }
}

// ---------------------------------------------------------------------------
extern "C" void kernel_launch(void* const* d_in, const int* in_sizes, int n_in,
                              void* d_out, int out_size, void* d_ws, size_t ws_size,
                              hipStream_t stream)
{
    (void)in_sizes; (void)n_in; (void)out_size; (void)ws_size;
    const int Bn = 8, T = 1024, F = 512, H = 2048;
    const int R = Bn * T;                      // 8192 token rows

    const float* x      = (const float*)d_in[0];
    const float* wq_w   = (const float*)d_in[1];
    const float* wq_b   = (const float*)d_in[2];
    const float* wk_w   = (const float*)d_in[3];
    const float* wk_b   = (const float*)d_in[4];
    const float* wv_w   = (const float*)d_in[5];
    const float* wv_b   = (const float*)d_in[6];
    const float* w_pos  = (const float*)d_in[7];
    const float* out_w  = (const float*)d_in[8];
    const float* out_b  = (const float*)d_in[9];
    const float* ln1_g  = (const float*)d_in[10];
    const float* ln1_b  = (const float*)d_in[11];
    const float* ln2_g  = (const float*)d_in[12];
    const float* ln2_b  = (const float*)d_in[13];
    const float* mlp1_w = (const float*)d_in[14];
    const float* mlp1_b = (const float*)d_in[15];
    const float* mlp2_w = (const float*)d_in[16];
    const float* mlp2_b = (const float*)d_in[17];

    char* ws = (char*)d_ws;
    size_t off = 0;
    auto take = [&](size_t bytes) -> char* {
        char* p = ws + off;
        off += (bytes + 255) & ~(size_t)255;
        return p;
    };

    __bf16* WTQ  = (__bf16*)take((size_t)F * F * 2);
    __bf16* WTK  = (__bf16*)take((size_t)F * F * 2);
    __bf16* WTV  = (__bf16*)take((size_t)F * F * 2);
    __bf16* WTO  = (__bf16*)take((size_t)F * F * 2);
    __bf16* WTM1 = (__bf16*)take((size_t)F * H * 2);
    __bf16* WTM2 = (__bf16*)take((size_t)H * F * 2);
    __bf16* EXPW = (__bf16*)take((size_t)T * T * 2);
    __bf16* ACTB = (__bf16*)take((size_t)R * F * 2);   // h -> Yt -> m (reused)
    float*  Qb   = (float*) take((size_t)R * F * 4);   // Q -> sigQ -> residual out
    float*  Kb   = (float*) take((size_t)R * F * 4);
    float*  Vb   = (float*) take((size_t)R * F * 4);
    __bf16* ATTB = (__bf16*)take((size_t)Bn * 2 * F * T * 2);
    float*  ATTC = (float*) take((size_t)Bn * T * 2 * F * 4);
    __bf16* T1B  = (__bf16*)take((size_t)R * H * 2);

    const dim3 blk(256);

    // ---- one-time prep: bf16-transposed weights + stabilized exp_w --------
    k_wt_bf16<<<(F * F + 255) / 256, blk, 0, stream>>>(wq_w,   WTQ,  F, F);
    k_wt_bf16<<<(F * F + 255) / 256, blk, 0, stream>>>(wk_w,   WTK,  F, F);
    k_wt_bf16<<<(F * F + 255) / 256, blk, 0, stream>>>(wv_w,   WTV,  F, F);
    k_wt_bf16<<<(F * F + 255) / 256, blk, 0, stream>>>(out_w,  WTO,  F, F);
    k_wt_bf16<<<(F * H + 255) / 256, blk, 0, stream>>>(mlp1_w, WTM1, F, H);
    k_wt_bf16<<<(H * F + 255) / 256, blk, 0, stream>>>(mlp2_w, WTM2, H, F);
    k_expw<<<T, blk, 0, stream>>>(w_pos, EXPW, T);

    // ---- LN1 -> h (bf16) --------------------------------------------------
    k_ln<<<R, blk, 0, stream>>>(x, ln1_g, ln1_b, ACTB, F);

    // ---- Q,K,V projections (WMMA) ----------------------------------------
    const dim3 gP(F / BN, R / BM, 1);
    k_gemm_bf16<<<gP, blk, 0, stream>>>(ACTB, 0, F, WTQ, 0, F, wq_b,
                                        Qb, 0, nullptr, 0, nullptr, 0, F, F, 0);
    k_gemm_bf16<<<gP, blk, 0, stream>>>(ACTB, 0, F, WTK, 0, F, wk_b,
                                        Kb, 0, nullptr, 0, nullptr, 0, F, F, 0);
    k_gemm_bf16<<<gP, blk, 0, stream>>>(ACTB, 0, F, WTV, 0, F, wv_b,
                                        Vb, 0, nullptr, 0, nullptr, 0, F, F, 0);

    // ---- AFT epilogue prep + fused num/den batched GEMM -------------------
    k_qkv_epi<<<R, blk, 0, stream>>>(Qb, Kb, Vb, ATTB, F, T);
    const dim3 gA((2 * F) / BN, T / BM, Bn);
    k_gemm_bf16<<<gA, blk, 0, stream>>>(EXPW, 0, T,
                                        ATTB, (long long)2 * F * T, T, nullptr,
                                        ATTC, (long long)T * 2 * F,
                                        nullptr, 0, nullptr, 0, 2 * F, T, 0);
    k_yt<<<R, blk, 0, stream>>>(Qb, ATTC, ACTB /*Yt*/, F, T);

    // ---- out projection + residual (C = Yt@Wo + b + x) --------------------
    k_gemm_bf16<<<gP, blk, 0, stream>>>(ACTB, 0, F, WTO, 0, F, out_b,
                                        Qb /*out*/, 0, x, 0, nullptr, 0, F, F, 0);

    // ---- LN2 -> m (bf16), MLP with fused GELU epilogues -------------------
    k_ln<<<R, blk, 0, stream>>>(Qb, ln2_g, ln2_b, ACTB, F);
    const dim3 gM1(H / BN, R / BM, 1);
    k_gemm_bf16<<<gM1, blk, 0, stream>>>(ACTB, 0, F, WTM1, 0, F, mlp1_b,
                                         nullptr, 0, nullptr, 0,
                                         T1B, 0, H, F, 1);
    const dim3 gM2(F / BN, R / BM, 1);
    k_gemm_bf16<<<gM2, blk, 0, stream>>>(T1B, 0, H, WTM2, 0, H, mlp2_b,
                                         (float*)d_out, 0, Qb /*residual*/, 0,
                                         nullptr, 0, F, H, 1);
}
